// GRUAnomalyDetector_8753143349749
// MI455X (gfx1250) — matrix operand order repaired and checked by reading, not compiled, not run on hardware
//
#include <hip/hip_runtime.h>
#include <hip/hip_bf16.h>

typedef __attribute__((ext_vector_type(16))) __bf16 v16bf;
typedef __attribute__((ext_vector_type(8)))  __bf16 v8bf;
typedef __attribute__((ext_vector_type(8)))  float  v8f;
typedef __attribute__((ext_vector_type(2)))  float  v2f;

#define Bb 2048
#define Tt 256
#define Ff 128
#define Hh 64
#define XPITCH 136   // bf16 elements per padded x row (128 + 8)
#define HPITCH 68    // fp32 elements per padded h row (64 + 4): full 64-bank tiling

#ifndef __has_builtin
#define __has_builtin(x) 0
#endif

__device__ __forceinline__ float fast_tanh(float x) {
#if __has_builtin(__builtin_amdgcn_tanhf)
    return __builtin_amdgcn_tanhf(x);   // v_tanh_f32 (gfx1250 TRANS op)
#else
    return tanhf(x);
#endif
}

__device__ __forceinline__ float sigmf(float x) {
    // sigmoid(x) = 0.5*tanh(x/2) + 0.5  -> mul + tanh + fma
    return __builtin_fmaf(0.5f, fast_tanh(0.5f * x), 0.5f);
}

// Pack 16 fp32 (4 float4) into 16 bf16 (2 x v8bf).
__device__ __forceinline__ void pack16(const float4& a, const float4& b,
                                       const float4& c, const float4& d,
                                       v8bf& q0, v8bf& q1) {
    q0[0] = (__bf16)a.x; q0[1] = (__bf16)a.y; q0[2] = (__bf16)a.z; q0[3] = (__bf16)a.w;
    q0[4] = (__bf16)b.x; q0[5] = (__bf16)b.y; q0[6] = (__bf16)b.z; q0[7] = (__bf16)b.w;
    q1[0] = (__bf16)c.x; q1[1] = (__bf16)c.y; q1[2] = (__bf16)c.z; q1[3] = (__bf16)c.w;
    q1[4] = (__bf16)d.x; q1[5] = (__bf16)d.y; q1[6] = (__bf16)d.z; q1[7] = (__bf16)d.w;
}

// A-fragment (16-bit 16x32 layout) read: dwords v=0..3 are 8 consecutive bf16
// at k = kb, v=4..7 at k = kb+16 -> two 16B LDS loads.
__device__ __forceinline__ v16bf load_xfrag(const __bf16* rowbase, int kb) {
    v8bf lo = *(const v8bf*)(rowbase + kb);
    v8bf hi = *(const v8bf*)(rowbase + kb + 16);
    return __builtin_shufflevector(lo, hi, 0, 1, 2, 3, 4, 5, 6, 7,
                                           8, 9, 10, 11, 12, 13, 14, 15);
}

// One workgroup = 16 batches, 4 waves. Wave w owns hidden columns [16w,16w+16)
// for all three gates. Weight B-fragments live in registers the whole T loop
// (bf16 wih: 96 VGPRs, f32 whh: 96 VGPRs). x tiles double-buffered in LDS as
// bf16 (converted once by the cooperative loader); h shared via [m][k] LDS
// with pitch 68 so f32 A-fragments are single ds_load_b64, conflict-free.
__launch_bounds__(128, 1)
__global__ void gru_fused_kernel(const float* __restrict__ x,
                                 const float* __restrict__ w_ih,
                                 const float* __restrict__ w_hh,
                                 const float* __restrict__ b_ih,
                                 const float* __restrict__ b_hh,
                                 const float* __restrict__ fc_w,
                                 const float* __restrict__ fc_b,
                                 float* __restrict__ out)
{
    __shared__ __bf16 xsb[2][16 * XPITCH]; // x tile in bf16, padded rows
    __shared__ float  hs[16 * HPITCH];     // h: [batch m][hidden k], pitch 68

    const int tid  = threadIdx.x;
    const int wave = tid >> 5;
    const int lane = tid & 31;
    const int lh   = lane >> 4;         // lane half (0/1)
    const int l16  = lane & 15;
    const int bb   = blockIdx.x * 16;   // batch tile base
    const int n0h  = wave * 16;         // hidden slice base for this wave

    // ---- persistent w_ih B-fragments (bf16, 32x16 per K-chunk) ----
    // B layout: kb = c*32 + lh*8; VGPRs 0-3 hold K=kb..kb+7, VGPRs 4-7 K=kb+16..kb+23
    // (N=lane), matching the A-fragment order.  B[k][n] = w_ih[n][k].
    v16bf wih[3][4];
    for (int g = 0; g < 3; ++g) {
        const float* wrow = w_ih + (size_t)(g * 64 + n0h + l16) * Ff;
        for (int c = 0; c < 4; ++c) {
            int kb = c * 32 + lh * 8;
            v16bf t;
            for (int v = 0; v < 8; ++v) {
                float2 p = *(const float2*)(wrow + kb + 2 * v + (v >> 2) * 8);
                t[2 * v]     = (__bf16)p.x;
                t[2 * v + 1] = (__bf16)p.y;
            }
            wih[g][c] = t;
        }
    }
    // ---- persistent w_hh B-fragments (f32, 4x16 per K-chunk) ----
    // B layout: lanes 0-15 -> K = 4q+v, lanes 16-31 -> K = 4q+2+v.
    v2f whh[3][16];
    for (int g = 0; g < 3; ++g) {
        const float* wrow = w_hh + (size_t)(g * 64 + n0h + l16) * Hh;
        for (int q = 0; q < 16; ++q) {
            float2 p = *(const float2*)(wrow + 4 * q + lh * 2);
            whh[g][q].x = p.x;
            whh[g][q].y = p.y;
        }
    }
    // biases as C/D-layout vectors (element (m,col) depends only on col)
    v8f biasR, biasZ, biasNX, biasNH;
    {
        const float br  = b_ih[n0h + l16]       + b_hh[n0h + l16];
        const float bz  = b_ih[64 + n0h + l16]  + b_hh[64 + n0h + l16];
        const float bnx = b_ih[128 + n0h + l16];
        const float bnh = b_hh[128 + n0h + l16];
        for (int i = 0; i < 8; ++i) {
            biasR[i] = br; biasZ[i] = bz; biasNX[i] = bnx; biasNH[i] = bnh;
        }
    }

    // h0 = 0
    for (int i = tid; i < 16 * HPITCH; i += 128) hs[i] = 0.0f;

    // cooperative x loader mapping: thread -> (row cm, 16 cols at ck)
    const int cm = tid >> 3;
    const int ck = (tid & 7) * 16;
    const float* xrow = x + (size_t)(bb + cm) * (Tt * Ff) + ck;

    {   // x[:, 0, :] -> bf16 LDS
        float4 a = *(const float4*)(xrow + 0);
        float4 b = *(const float4*)(xrow + 4);
        float4 c = *(const float4*)(xrow + 8);
        float4 d = *(const float4*)(xrow + 12);
        v8bf q0, q1;
        pack16(a, b, c, d, q0, q1);
        __bf16* dst = &xsb[0][cm * XPITCH + ck];
        *(v8bf*)(dst + 0) = q0;
        *(v8bf*)(dst + 8) = q1;
    }
    __syncthreads();

    for (int t = 0; t < Tt; ++t) {
        const int cur = t & 1;

        // keep the x stream ahead of the blocking loads
        if (t + 4 < Tt) __builtin_prefetch(xrow + (size_t)(t + 4) * Ff, 0, 3);

        // issue prefetch of x[:, t+1, :] early; consumed after the WMMA work
        float4 p0, p1, p2, p3;
        const bool pf = (t + 1 < Tt);
        if (pf) {
            const float* p = xrow + (size_t)(t + 1) * Ff;
            p0 = *(const float4*)(p + 0);
            p1 = *(const float4*)(p + 4);
            p2 = *(const float4*)(p + 8);
            p3 = *(const float4*)(p + 12);
        }

        // ---- gx contribution: bf16 WMMA, K = 4 x 32 ----
        // chunk 0 takes the bias vectors directly as C (no accumulator init)
        const __bf16* xb = &xsb[cur][l16 * XPITCH];
        v8f accR, accZ, accNX, accNH;
        {
            v16bf xa = load_xfrag(xb, lh * 8);
            accR  = __builtin_amdgcn_wmma_f32_16x16x32_bf16(false, xa, false, wih[0][0], (short)0, biasR,  false, false);
            accZ  = __builtin_amdgcn_wmma_f32_16x16x32_bf16(false, xa, false, wih[1][0], (short)0, biasZ,  false, false);
            accNX = __builtin_amdgcn_wmma_f32_16x16x32_bf16(false, xa, false, wih[2][0], (short)0, biasNX, false, false);
        }
        for (int c = 1; c < 4; ++c) {
            v16bf xa = load_xfrag(xb, c * 32 + lh * 8);
            accR  = __builtin_amdgcn_wmma_f32_16x16x32_bf16(false, xa, false, wih[0][c], (short)0, accR,  false, false);
            accZ  = __builtin_amdgcn_wmma_f32_16x16x32_bf16(false, xa, false, wih[1][c], (short)0, accZ,  false, false);
            accNX = __builtin_amdgcn_wmma_f32_16x16x32_bf16(false, xa, false, wih[2][c], (short)0, accNX, false, false);
        }

        // ---- gh contribution: f32 WMMA, K = 16 x 4 ----
        // A layout (f32 16x4): lane m = l16, VGPR v -> K = 4q + lh*2 + v.
        // [m][k] pitch-68 LDS makes each fragment one even-aligned ds_load_b64.
        const float* hrow = &hs[l16 * HPITCH + lh * 2];
        {
            v2f ha = *(const v2f*)(hrow + 0);
            accR  = __builtin_amdgcn_wmma_f32_16x16x4_f32(false, ha, false, whh[0][0], (short)0, accR,   false, false);
            accZ  = __builtin_amdgcn_wmma_f32_16x16x4_f32(false, ha, false, whh[1][0], (short)0, accZ,   false, false);
            accNH = __builtin_amdgcn_wmma_f32_16x16x4_f32(false, ha, false, whh[2][0], (short)0, biasNH, false, false);
        }
        for (int q = 1; q < 16; ++q) {
            v2f ha = *(const v2f*)(hrow + 4 * q);
            accR  = __builtin_amdgcn_wmma_f32_16x16x4_f32(false, ha, false, whh[0][q], (short)0, accR,  false, false);
            accZ  = __builtin_amdgcn_wmma_f32_16x16x4_f32(false, ha, false, whh[1][q], (short)0, accZ,  false, false);
            accNH = __builtin_amdgcn_wmma_f32_16x16x4_f32(false, ha, false, whh[2][q], (short)0, accNH, false, false);
        }

        // ---- gating in C/D layout: element (m = i or i+8, col = n0h+l16) ----
        float hnew[8];
        for (int i = 0; i < 8; ++i) {
            int m = lh ? (i + 8) : i;
            float hold = hs[m * HPITCH + n0h + l16];
            float r = sigmf(accR[i]);
            float z = sigmf(accZ[i]);
            float n = fast_tanh(__builtin_fmaf(r, accNH[i], accNX[i]));
            hnew[i] = (1.0f - z) * n + z * hold;
        }

        __syncthreads();  // all waves done reading hs / xsb[cur]
        for (int i = 0; i < 8; ++i) {
            int m = lh ? (i + 8) : i;
            hs[m * HPITCH + n0h + l16] = hnew[i];
        }
        if (pf) {
            v8bf q0, q1;
            pack16(p0, p1, p2, p3, q0, q1);
            __bf16* dst = &xsb[cur ^ 1][cm * XPITCH + ck];
            *(v8bf*)(dst + 0) = q0;
            *(v8bf*)(dst + 8) = q1;
        }
        __syncthreads();  // writes visible before next step's reads
    }

    // final FC: out[b] = h_last[b,:] . fc_w + fc_b
    if (tid < 16) {
        float acc = fc_b[0];
        for (int k = 0; k < Hh; ++k) acc += hs[tid * HPITCH + k] * fc_w[k];
        out[bb + tid] = acc;
    }
}

extern "C" void kernel_launch(void* const* d_in, const int* in_sizes, int n_in,
                              void* d_out, int out_size, void* d_ws, size_t ws_size,
                              hipStream_t stream) {
    const float* x    = (const float*)d_in[0];
    const float* w_ih = (const float*)d_in[1];
    const float* w_hh = (const float*)d_in[2];
    const float* b_ih = (const float*)d_in[3];
    const float* b_hh = (const float*)d_in[4];
    const float* fc_w = (const float*)d_in[5];
    const float* fc_b = (const float*)d_in[6];
    float* out = (float*)d_out;

    dim3 grid(Bb / 16), block(128);
    hipLaunchKernelGGL(gru_fused_kernel, grid, block, 0, stream,
                       x, w_ih, w_hh, b_ih, b_hh, fc_w, fc_b, out);
    (void)in_sizes; (void)n_in; (void)out_size; (void)d_ws; (void)ws_size;
}
